// Model_75694503624892
// MI455X (gfx1250) — matrix operand order repaired and checked by reading, not compiled, not run on hardware
//
#include <hip/hip_runtime.h>
#include <math.h>

typedef unsigned short u16;
typedef __bf16 bf16_t;
typedef bf16_t v16bf __attribute__((ext_vector_type(16)));
typedef float  v8f  __attribute__((ext_vector_type(8)));
typedef int    v4i  __attribute__((ext_vector_type(4)));

#define TILE_BM 128
#define TILE_BN 128
#define TILE_BK 32
#define LDS_STR 40   // padded LDS row stride in shorts (80B -> 16B-aligned rows)

#if __has_builtin(__builtin_amdgcn_global_load_async_to_lds_b128) && \
    __has_builtin(__builtin_amdgcn_s_wait_asynccnt)
#define USE_ASYNC_LDS 1
#else
#define USE_ASYNC_LDS 0
#endif

__device__ __forceinline__ u16 f32_to_bf16(float f) {
  unsigned int u = __float_as_uint(f);
  if ((u & 0x7F800000u) == 0x7F800000u) return (u16)(u >> 16); // inf/nan passthrough
  unsigned int r = u + 0x7FFFu + ((u >> 16) & 1u);             // round-to-nearest-even
  return (u16)(r >> 16);
}

// copy 16 consecutive bf16 (one thread's share of a tile row) global -> LDS
__device__ __forceinline__ void stage16(const u16* __restrict__ srcRow, bool rowOk,
                                        int gk, int Kv, u16* dst) {
  if (rowOk && gk + 16 <= Kv) {
#if USE_ASYNC_LDS
    __builtin_amdgcn_global_load_async_to_lds_b128((v4i*)(srcRow + gk), (v4i*)dst, 0, 0);
    __builtin_amdgcn_global_load_async_to_lds_b128((v4i*)(srcRow + gk + 8), (v4i*)(dst + 8), 0, 0);
#else
    const uint4* s4 = (const uint4*)(srcRow + gk);
    uint4 v0 = s4[0], v1 = s4[1];
    *(uint4*)dst = v0;
    *(uint4*)(dst + 8) = v1;
#endif
  } else {
#pragma unroll 1
    for (int i = 0; i < 16; ++i)
      dst[i] = (rowOk && (gk + i) < Kv) ? srcRow[gk + i] : (u16)0;
  }
}

struct GemmP {
  const u16* A;        // bf16 [M,K] row-major (lda), batch stride sA
  const u16* Bm;       // bf16 [N,K] row-major (ldb), batch stride sB
  float* Cf;           // fp32 out (or null)
  u16* Cb;             // bf16 out (or null)
  const float* bias;   // fp32 bias[n] (or null)
  long long lda, ldb, ldc, sA, sB, sC;
  int M, N, Kv;        // Kv = valid K extent (loads beyond read as 0)
  int transC, accum;   // transC: bf16 store at [n,m]; accum: fp32 C += result
  float alpha;
};

union Frag { uint4 q[2]; v16bf v; };

__global__ __launch_bounds__(256) void gemm_bf16_wmma(GemmP p) {
  __shared__ __align__(16) u16 lA[2][TILE_BM * LDS_STR];
  __shared__ __align__(16) u16 lB[2][TILE_BN * LDS_STR];

  const int tid = threadIdx.x;
  const int m0 = blockIdx.y * TILE_BM;
  const int n0 = blockIdx.x * TILE_BN;
  const size_t z = blockIdx.z;
  const u16* __restrict__ A  = p.A  + z * (size_t)p.sA;
  const u16* __restrict__ Bp = p.Bm + z * (size_t)p.sB;

  const int lane = tid & 31;
  const int wid  = tid >> 5;
  const int wm   = (wid >> 2) * 64;   // wave tile origin in M (0 or 64)
  const int wn   = (wid & 3) * 32;    // wave tile origin in N (0..96)
  const int lm   = lane & 15;
  const int hi   = lane >> 4;

  // staging role: thread copies 16 consecutive shorts of tile row `row`
  const int row = tid >> 1;
  const int kb  = (tid & 1) * 16;
  const bool aOk = (m0 + row) < p.M;
  const bool bOk = (n0 + row) < p.N;
  const u16* aRow = A  + (size_t)(m0 + row) * p.lda;
  const u16* bRow = Bp + (size_t)(n0 + row) * p.ldb;
  const int ldsOff = row * LDS_STR + kb;

  v8f acc[4][2];
  for (int i = 0; i < 4; ++i)
    for (int j = 0; j < 2; ++j)
      for (int r = 0; r < 8; ++r)
        acc[i][j][r] = 0.0f;

  const int ksteps = (p.Kv + TILE_BK - 1) / TILE_BK;

  // prologue: stage tile 0 into buffer 0
  stage16(aRow, aOk, kb, p.Kv, &lA[0][ldsOff]);
  stage16(bRow, bOk, kb, p.Kv, &lB[0][ldsOff]);
#if USE_ASYNC_LDS
  __builtin_amdgcn_s_wait_asynccnt(0);
#endif
  __syncthreads();

  for (int kt = 0; kt < ksteps; ++kt) {
    const int cur = kt & 1;
    const bool more = (kt + 1) < ksteps;

    if (more) {
      const int gk = (kt + 1) * TILE_BK + kb;
      stage16(aRow, aOk, gk, p.Kv, &lA[1 - cur][ldsOff]);
      stage16(bRow, bOk, gk, p.Kv, &lB[1 - cur][ldsOff]);
      if (kt + 2 < ksteps) {
        if (aOk) __builtin_prefetch(aRow + (kt + 2) * TILE_BK + kb, 0, 3);
        if (bOk) __builtin_prefetch(bRow + (kt + 2) * TILE_BK + kb, 0, 3);
      }
    }

    // ---- fragments per documented wave32 VGPR layouts ----
    Frag a[4], b[2];
#pragma unroll
    for (int i = 0; i < 4; ++i) {
      const u16* rp = &lA[cur][(wm + i * 16 + lm) * LDS_STR];
      a[i].q[0] = *(const uint4*)(rp + 8 * hi);        // K = 8*hi .. 8*hi+7
      a[i].q[1] = *(const uint4*)(rp + 16 + 8 * hi);   // K = 16+8*hi ..
    }
#pragma unroll
    for (int j = 0; j < 2; ++j) {
      const u16* cp = &lB[cur][(wn + j * 16 + lm) * LDS_STR + 8 * hi];
      b[j].q[0] = *(const uint4*)(cp);                 // K = 8*hi .. 8*hi+7
      b[j].q[1] = *(const uint4*)(cp + 16);            // K = 16+8*hi ..
    }

#pragma unroll
    for (int i = 0; i < 4; ++i)
#pragma unroll
      for (int j = 0; j < 2; ++j)
        acc[i][j] = __builtin_amdgcn_wmma_f32_16x16x32_bf16(
            false, a[i].v, false, b[j].v, (short)0, acc[i][j], false, false);

#if USE_ASYNC_LDS
    if (more) __builtin_amdgcn_s_wait_asynccnt(0);
#endif
    __syncthreads();
  }

  // ---- epilogue: alpha, bias, optional f32 accumulate, f32 / (transposed) bf16 store ----
#pragma unroll
  for (int i = 0; i < 4; ++i) {
#pragma unroll
    for (int j = 0; j < 2; ++j) {
      const int gn = n0 + wn + j * 16 + lm;
      if (gn >= p.N) continue;
      const float bv = p.bias ? p.bias[gn] : 0.0f;
#pragma unroll
      for (int r = 0; r < 8; ++r) {
        const int gm = m0 + wm + i * 16 + r + 8 * hi;
        if (gm >= p.M) continue;
        float v = p.alpha * acc[i][j][r] + bv;
        if (p.Cf) {
          const size_t idx = z * (size_t)p.sC + (size_t)gm * p.ldc + gn;
          if (p.accum) v += p.Cf[idx];
          p.Cf[idx] = v;
        } else {
          const size_t idx = z * (size_t)p.sC +
                             (p.transC ? (size_t)gn * p.ldc + gm
                                       : (size_t)gm * p.ldc + gn);
          p.Cb[idx] = f32_to_bf16(v);
        }
      }
    }
  }
}

// ---- forward DFT matrices: Wc[f,t]=cos(2pi f t/T)/sqrt(T), Ws=-sin(.)/sqrt(T) ----
__global__ void fill_fwd_dft(u16* Wc, u16* Ws, int F, int T, float rs) {
  size_t idx = (size_t)blockIdx.x * blockDim.x + threadIdx.x;
  size_t total = (size_t)F * T;
  if (idx >= total) return;
  int f = (int)(idx / T), t = (int)(idx % T);
  long long m = ((long long)f * t) % T;
  float ang = (float)m * (6.283185307179586f / (float)T);
  float s, c;
  sincosf(ang, &s, &c);
  Wc[idx] = f32_to_bf16(c * rs);
  Ws[idx] = f32_to_bf16(-s * rs);
}

// ---- inverse (ortho irfft) matrices, padded to Fpad cols with zeros ----
__global__ void fill_inv_dft(u16* Vc, u16* Vs, int T, int F, int Fpad, float rs) {
  size_t idx = (size_t)blockIdx.x * blockDim.x + threadIdx.x;
  size_t total = (size_t)T * Fpad;
  if (idx >= total) return;
  int t = (int)(idx / Fpad), f = (int)(idx % Fpad);
  if (f >= F) { Vc[idx] = 0; Vs[idx] = 0; return; }
  float w = (f == 0 || f == T / 2) ? 1.0f : 2.0f;   // Hermitian fold weight
  long long m = ((long long)f * t) % T;
  float ang = (float)m * (6.283185307179586f / (float)T);
  float s, c;
  sincosf(ang, &s, &c);
  Vc[idx] = f32_to_bf16(w * c * rs);
  Vs[idx] = f32_to_bf16(-w * s * rs);
}

__global__ void cvt_f32_bf16(const float* __restrict__ in, u16* __restrict__ out, size_t n) {
  size_t i = (size_t)blockIdx.x * blockDim.x + threadIdx.x;
  if (i < n) out[i] = f32_to_bf16(in[i]);
}

// LDS-tiled transpose + fp32->bf16: in [R,C] f32 per batch -> out [C,R] bf16
__global__ __launch_bounds__(256) void transpose_cvt(const float* __restrict__ in,
                                                     u16* __restrict__ out,
                                                     int R, int C,
                                                     long long sIn, long long sOut) {
  __shared__ float tile[32][33];
  const int tx = threadIdx.x, ty = threadIdx.y;   // block (32, 8)
  const size_t z = blockIdx.z;
  const int r0 = blockIdx.y * 32, c0 = blockIdx.x * 32;
  const float* src = in + z * (size_t)sIn;
  u16* dst = out + z * (size_t)sOut;
#pragma unroll
  for (int i = 0; i < 4; ++i) {
    int r = r0 + ty + i * 8, c = c0 + tx;
    tile[ty + i * 8][tx] = (r < R && c < C) ? src[(size_t)r * C + c] : 0.0f;
  }
  __syncthreads();
#pragma unroll
  for (int i = 0; i < 4; ++i) {
    int c = c0 + ty + i * 8, r = r0 + tx;
    if (c < C && r < R) dst[(size_t)c * R + r] = f32_to_bf16(tile[tx][ty + i * 8]);
  }
}

// one block per (b,f) row: softmax over F cols, hard-sparsify, write bf16 padded row
__global__ __launch_bounds__(256) void softmax_sparsify(const float* __restrict__ S,
                                                        u16* __restrict__ P,
                                                        int F, int Fpad) {
  const size_t row = blockIdx.x;
  const float* s = S + row * (size_t)F;
  u16* p = P + row * (size_t)Fpad;
  const int tid = threadIdx.x;
  __shared__ float red[8];

  float lmax = -3.4e38f;
  for (int g = tid; g < F; g += 256) lmax = fmaxf(lmax, s[g]);
  for (int o = 16; o; o >>= 1) lmax = fmaxf(lmax, __shfl_xor(lmax, o, 32));
  if ((tid & 31) == 0) red[tid >> 5] = lmax;
  __syncthreads();
  float bmax = red[0];
  for (int i = 1; i < 8; ++i) bmax = fmaxf(bmax, red[i]);
  __syncthreads();

  float lsum = 0.0f;
  for (int g = tid; g < F; g += 256) lsum += __expf(s[g] - bmax);
  for (int o = 16; o; o >>= 1) lsum += __shfl_xor(lsum, o, 32);
  if ((tid & 31) == 0) red[tid >> 5] = lsum;
  __syncthreads();
  float bsum = 0.0f;
  for (int i = 0; i < 8; ++i) bsum += red[i];
  const float inv = 1.0f / bsum;

  for (int g = tid; g < Fpad; g += 256) {
    float v = 0.0f;
    if (g < F) {
      float pr = __expf(s[g] - bmax) * inv;
      v = (pr > 0.01f) ? pr : 0.0f;   // SPARSITY_THRESHOLD
    }
    p[g] = f32_to_bf16(v);
  }
}

extern "C" void kernel_launch(void* const* d_in, const int* in_sizes, int n_in,
                              void* d_out, int out_size, void* d_ws, size_t ws_size,
                              hipStream_t stream) {
  (void)in_sizes; (void)n_in; (void)out_size; (void)ws_size;
  constexpr int NB = 16, T = 2048, D = 512, F = 1025, Fpad = 1040;
  typedef long long ll;

  const float* x1 = (const float*)d_in[0];
  const float* x2 = (const float*)d_in[1];
  const float* w_f32[8] = { (const float*)d_in[2],  (const float*)d_in[4],
                            (const float*)d_in[6],  (const float*)d_in[8],
                            (const float*)d_in[10], (const float*)d_in[12],
                            (const float*)d_in[14], (const float*)d_in[16] };
  const float* rq_b = (const float*)d_in[3];
  const float* rk_b = (const float*)d_in[5];
  const float* rv_b = (const float*)d_in[7];
  const float* iq_b = (const float*)d_in[9];
  const float* ik_b = (const float*)d_in[11];
  const float* iv_b = (const float*)d_in[13];
  const float* or_b = (const float*)d_in[15];
  const float* oi_b = (const float*)d_in[17];

  char* ws = (char*)d_ws;
  size_t off = 0;
  auto take = [&](size_t bytes) -> char* {
    char* pp = ws + off;
    off = (off + bytes + 255) & ~(size_t)255;
    return pp;
  };

  u16* Wc  = (u16*)take((size_t)F * T * 2);
  u16* Ws_ = (u16*)take((size_t)F * T * 2);
  u16* Vc  = (u16*)take((size_t)T * Fpad * 2);
  u16* Vs  = (u16*)take((size_t)T * Fpad * 2);
  u16* x1t = (u16*)take((size_t)NB * D * T * 2);   // [b][d][t]
  u16* x2t = (u16*)take((size_t)NB * D * T * 2);
  u16* wb[8];
  for (int i = 0; i < 8; ++i) wb[i] = (u16*)take((size_t)D * D * 2);
  const size_t nBFD = (size_t)NB * F * D;
  const size_t nBDFp = (size_t)NB * D * Fpad;
  u16* r1 = (u16*)take(nBFD * 2);   u16* i1 = (u16*)take(nBFD * 2);
  u16* r2 = (u16*)take(nBFD * 2);   u16* i2 = (u16*)take(nBFD * 2);
  u16* rQ = (u16*)take(nBFD * 2);   u16* iQ = (u16*)take(nBFD * 2);
  u16* rK = (u16*)take(nBFD * 2);   u16* iK = (u16*)take(nBFD * 2);
  u16* rVt = (u16*)take(nBDFp * 2); u16* iVt = (u16*)take(nBDFp * 2);   // [b][d][fpad]
  u16* ctxr = (u16*)take(nBFD * 2); u16* ctxi = (u16*)take(nBFD * 2);
  u16* outrt = (u16*)take(nBDFp * 2); u16* outit = (u16*)take(nBDFp * 2);
  float* scores = (float*)take((size_t)NB * F * F * 4);
  u16* attnb = (u16*)take((size_t)NB * F * Fpad * 2);

  const float rs = 1.0f / sqrtf((float)T);
  const float scale = 0.125f;  // (D/NUM_HEADS)^-0.5 = 64^-0.5

  auto gemm = [&](const u16* Am, ll lda, ll sA,
                  const u16* Bmat, ll ldb, ll sB,
                  float* Cf, u16* Cb, ll ldc, ll sC, bool transC,
                  const float* bias, float alpha, bool accum,
                  int M, int N, int Kv) {
    GemmP gp{Am, Bmat, Cf, Cb, bias, lda, ldb, ldc, sA, sB, sC,
             M, N, Kv, transC ? 1 : 0, accum ? 1 : 0, alpha};
    dim3 grid((N + TILE_BN - 1) / TILE_BN, (M + TILE_BM - 1) / TILE_BM, NB);
    hipLaunchKernelGGL(gemm_bf16_wmma, grid, dim3(256), 0, stream, gp);
  };

  // DFT / iDFT matrices
  {
    size_t n = (size_t)F * T;
    hipLaunchKernelGGL(fill_fwd_dft, dim3((n + 255) / 256), dim3(256), 0, stream,
                       Wc, Ws_, F, T, rs);
    size_t m = (size_t)T * Fpad;
    hipLaunchKernelGGL(fill_inv_dft, dim3((m + 255) / 256), dim3(256), 0, stream,
                       Vc, Vs, T, F, Fpad, rs);
  }
  // x -> bf16, transposed to [b][d][t]; weights -> bf16 (already [N,K])
  {
    dim3 tg((D + 31) / 32, (T + 31) / 32, NB);
    hipLaunchKernelGGL(transpose_cvt, tg, dim3(32, 8), 0, stream,
                       x1, x1t, T, D, (ll)T * D, (ll)D * T);
    hipLaunchKernelGGL(transpose_cvt, tg, dim3(32, 8), 0, stream,
                       x2, x2t, T, D, (ll)T * D, (ll)D * T);
    size_t nw = (size_t)D * D;
    for (int i = 0; i < 8; ++i)
      hipLaunchKernelGGL(cvt_f32_bf16, dim3((nw + 255) / 256), dim3(256), 0, stream,
                         w_f32[i], wb[i], nw);
  }

  // rfft via GEMM: r/i[b,f,d] = sum_t W[f,t] * xT[b,d,t]
  gemm(Wc,  T, 0, x1t, T, (ll)D * T, nullptr, r1, D, (ll)F * D, false, nullptr, 1.f, false, F, D, T);
  gemm(Ws_, T, 0, x1t, T, (ll)D * T, nullptr, i1, D, (ll)F * D, false, nullptr, 1.f, false, F, D, T);
  gemm(Wc,  T, 0, x2t, T, (ll)D * T, nullptr, r2, D, (ll)F * D, false, nullptr, 1.f, false, F, D, T);
  gemm(Ws_, T, 0, x2t, T, (ll)D * T, nullptr, i2, D, (ll)F * D, false, nullptr, 1.f, false, F, D, T);

  // projections: x @ W^T + b (weights are [N,K] already); V stored transposed [d][fpad]
  gemm(r1, D, (ll)F * D, wb[0], D, 0, nullptr, rQ, D, (ll)F * D, false, rq_b, 1.f, false, F, D, D);
  gemm(r2, D, (ll)F * D, wb[1], D, 0, nullptr, rK, D, (ll)F * D, false, rk_b, 1.f, false, F, D, D);
  gemm(r2, D, (ll)F * D, wb[2], D, 0, nullptr, rVt, Fpad, (ll)D * Fpad, true, rv_b, 1.f, false, F, D, D);
  gemm(i1, D, (ll)F * D, wb[3], D, 0, nullptr, iQ, D, (ll)F * D, false, iq_b, 1.f, false, F, D, D);
  gemm(i2, D, (ll)F * D, wb[4], D, 0, nullptr, iK, D, (ll)F * D, false, ik_b, 1.f, false, F, D, D);
  gemm(i2, D, (ll)F * D, wb[5], D, 0, nullptr, iVt, Fpad, (ll)D * Fpad, true, iv_b, 1.f, false, F, D, D);

  // scores = scale * (rQ rK^T + iQ iK^T); K matrices are already [N,K]
  gemm(rQ, D, (ll)F * D, rK, D, (ll)F * D, scores, nullptr, F, (ll)F * F, false, nullptr, scale, false, F, F, D);
  gemm(iQ, D, (ll)F * D, iK, D, (ll)F * D, scores, nullptr, F, (ll)F * F, false, nullptr, scale, true,  F, F, D);

  // softmax + hard sparsity -> bf16 attn (padded cols zeroed)
  hipLaunchKernelGGL(softmax_sparsify, dim3(NB * F), dim3(256), 0, stream,
                     scores, attnb, F, Fpad);

  // ctx = attn @ V  (V held transposed: B = Vt [d][fpad] = [N,K])
  gemm(attnb, Fpad, (ll)F * Fpad, rVt, Fpad, (ll)D * Fpad, nullptr, ctxr, D, (ll)F * D, false, nullptr, 1.f, false, F, D, F);
  gemm(attnb, Fpad, (ll)F * Fpad, iVt, Fpad, (ll)D * Fpad, nullptr, ctxi, D, (ll)F * D, false, nullptr, 1.f, false, F, D, F);

  // output projections, stored transposed [d][fpad] for the iDFT's B operand
  gemm(ctxr, D, (ll)F * D, wb[6], D, 0, nullptr, outrt, Fpad, (ll)D * Fpad, true, or_b, 1.f, false, F, D, D);
  gemm(ctxi, D, (ll)F * D, wb[7], D, 0, nullptr, outit, Fpad, (ll)D * Fpad, true, oi_b, 1.f, false, F, D, D);

  // irfft via GEMM: y[b,t,d] = sum_f Vc[t,f]*outR[f,d] + Vs[t,f]*outI[f,d]
  gemm(Vc, Fpad, 0, outrt, Fpad, (ll)D * Fpad, (float*)d_out, nullptr, D, (ll)T * D, false, nullptr, 1.f, false, T, D, F);
  gemm(Vs, Fpad, 0, outit, Fpad, (ll)D * Fpad, (float*)d_out, nullptr, D, (ll)T * D, false, nullptr, 1.f, true,  T, D, F);
}
